// MyDCNv2_88201448391342
// MI455X (gfx1250) — matrix operand order repaired and checked by reading, not compiled, not run on hardware
//
#include <hip/hip_runtime.h>

typedef __attribute__((ext_vector_type(16))) _Float16 v16h;
typedef __attribute__((ext_vector_type(8)))  float    v8f;

#define C_IN   256
#define OUTC   256
#define HDIM   64
#define WDIM   64
#define KTAPS  9
#define KDIM   (C_IN * KTAPS)      // 2304
#define KCH    32                  // WMMA K per step
#define NCHUNK (KDIM / KCH)        // 72
#define PXW    64                  // pixels per workgroup (one image row)

// packed A-fragment weights: [kc][m(16)][lane(32)][8 dwords] = 1.18 MB
#define WS_DWORDS ((size_t)NCHUNK * 16 * 32 * 8)
#define WS_BYTES  (WS_DWORDS * 4)

// ---------------------------------------------------------------------------
// One-time weight pre-pack: fp32 -> f16 in the exact CDNA5 16-bit A-matrix
// register layout, so the main loop loads A-fragments with 2x b128 directly.
// ---------------------------------------------------------------------------
__global__ __launch_bounds__(256)
void dcn_prepack_weight(const float* __restrict__ weight,
                        unsigned* __restrict__ wsw)
{
    int t = blockIdx.x * 256 + threadIdx.x;        // triple (kc, m, lane)
    if (t >= NCHUNK * 16 * 32) return;
    int lane  = t & 31;
    int m     = (t >> 5) & 15;
    int kc    = t >> 9;
    int arow  = m * 16 + (lane & 15);              // out channel
    int khalf = (lane >> 4) * 8;
    int i0    = kc * KCH;
    unsigned obase = (unsigned)t * 8u;
    #pragma unroll
    for (int d = 0; d < 8; ++d) {
        int K = (d < 4) ? (khalf + 2 * d) : (khalf + 16 + 2 * (d - 4));
        float w0 = weight[(size_t)arow * KDIM + i0 + K];
        float w1 = weight[(size_t)arow * KDIM + i0 + K + 1];
        union { _Float16 h[2]; unsigned u; } p;
        p.h[0] = (_Float16)w0;
        p.h[1] = (_Float16)w1;
        wsw[obase + d] = p.u;
    }
}

// ---------------------------------------------------------------------------
// Table-build: per (pixel, tap) compute 4 clamped corner BYTE offsets
// (uint4, load-ready) and 4 validity-folded bilinear*mask weights (float4).
// Inner-loop gathers become branchless, compare-free and ALU-free addressing.
// ---------------------------------------------------------------------------
__device__ __forceinline__
void build_table(int tid, int n, int h,
                 const float* __restrict__ offset,
                 const float* __restrict__ mask,
                 uint4*  __restrict__ tofs,
                 float4* __restrict__ twt4)
{
    for (int e = tid; e < PXW * KTAPS; e += 512) {
        int px = e / KTAPS;
        int k  = e - px * KTAPS;
        int kh = k / 3;
        int kw = k - kh * 3;
        float offy = offset[(((size_t)n * 18 + 2 * k    ) * HDIM + h) * WDIM + px];
        float offx = offset[(((size_t)n * 18 + 2 * k + 1) * HDIM + h) * WDIM + px];
        float m    = mask  [(((size_t)n * 9  + k        ) * HDIM + h) * WDIM + px];
        float py  = (float)(h  + kh) + offy;      // padded coords (ys+dy = h+kh)
        float pxf = (float)(px + kw) + offx;
        py  = fminf(fmaxf(py,  0.f), 65.f);
        pxf = fminf(fmaxf(pxf, 0.f), 65.f);
        float fy = floorf(py),  fx = floorf(pxf);
        float wy = py - fy,     wx = pxf - fx;
        int y0 = (int)fy, x0 = (int)fx;
        // real (unpadded) corner coords
        int ry0 = y0 - 1, rx0 = x0 - 1;
        int ry1 = y0,     rx1 = x0;               // (y0+1)-1
        // validity per corner (zero-padding emulation)
        float vy0 = ((unsigned)ry0 < 64u) ? 1.f : 0.f;
        float vy1 = ((unsigned)ry1 < 64u) ? 1.f : 0.f;
        float vx0 = ((unsigned)rx0 < 64u) ? 1.f : 0.f;
        float vx1 = ((unsigned)rx1 < 64u) ? 1.f : 0.f;
        // clamped in-range coords (loads are always safe)
        int cy0 = min(max(ry0, 0), 63), cx0 = min(max(rx0, 0), 63);
        int cy1 = min(max(ry1, 0), 63), cx1 = min(max(rx1, 0), 63);
        uint4 o;                                   // byte offsets within plane
        o.x = (unsigned)((cy0 * 64 + cx0) << 2);
        o.y = (unsigned)((cy0 * 64 + cx1) << 2);
        o.z = (unsigned)((cy1 * 64 + cx0) << 2);
        o.w = (unsigned)((cy1 * 64 + cx1) << 2);
        tofs[e] = o;
        float4 w;
        w.x = (1.f - wy) * (1.f - wx) * m * vy0 * vx0;
        w.y = (1.f - wy) * wx         * m * vy0 * vx1;
        w.z = wy         * (1.f - wx) * m * vy1 * vx0;
        w.w = wy         * wx         * m * vy1 * vx1;
        twt4[e] = w;
    }
}

// ---------------------------------------------------------------------------
// Main fused kernel: A-fragments streamed from packed global weights,
// im2col tile double-buffered in LDS (one barrier per K-iteration).
// ---------------------------------------------------------------------------
__global__ __launch_bounds__(512)
void dcnv2_fused_wmma_pk(const float* __restrict__ x,
                         const float* __restrict__ offset,
                         const float* __restrict__ mask,
                         const unsigned* __restrict__ wpack,
                         float* __restrict__ out)
{
    __shared__ _Float16 imtile[2][PXW * KCH];     // 2 x 4 KB double buffer
    __shared__ uint4    tofs[PXW * KTAPS];        // 9 KB corner byte-offsets
    __shared__ float4   twt4[PXW * KTAPS];        // 9 KB folded weights

    const int tid  = threadIdx.x;
    const int row  = blockIdx.x;          // (n, h)
    const int n    = row >> 6;
    const int h    = row & 63;
    const int lane = tid & 31;
    const int wv   = tid >> 5;            // wave -> out-channel tile

    build_table(tid, n, h, offset, mask, tofs, twt4);

    v8f acc[4];
    #pragma unroll
    for (int j = 0; j < 4; ++j)
        #pragma unroll
        for (int v = 0; v < 8; ++v) acc[j][v] = 0.f;

    const char* xbase = (const char*)(x + (size_t)n * C_IN * HDIM * WDIM);
    const int ipx = tid & 63;      // im2col pixel
    const int il0 = tid >> 6;      // im2col base i-local (0..7) -- wave-uniform

    // incremental (c, k) per j-slot: i = kc*32 + il0 + 8j, c = i/9, k = i%9
    int c4[4], k4[4];
    #pragma unroll
    for (int j = 0; j < 4; ++j) {
        int il = il0 + j * 8;
        c4[j] = il / 9;
        k4[j] = il - c4[j] * 9;
    }

    // stage chunk's im2col tile; advances (c,k) state by one chunk per call
    auto stage = [&](_Float16* buf) {
        #pragma unroll
        for (int j = 0; j < 4; ++j) {
            int il = il0 + j * 8;
            int c  = c4[j];
            int k  = k4[j];
            int e  = ipx * KTAPS + k;
            uint4  o = tofs[e];
            float4 w = twt4[e];
            // c is wave-uniform: force SGPR base so gathers use saddr+voffset
            int cbytes = __builtin_amdgcn_readfirstlane(c << 14); // c*64*64*4
            const char* xc = xbase + (unsigned)cbytes;
            float g00 = *(const float*)(xc + o.x);
            float g01 = *(const float*)(xc + o.y);
            float g10 = *(const float*)(xc + o.z);
            float g11 = *(const float*)(xc + o.w);
            float val = w.x * g00 + w.y * g01 + w.z * g10 + w.w * g11;
            buf[ipx * KCH + il] = (_Float16)val;
            // advance: i += 32  ->  c += 3, k += 5 (mod 9)
            k += 5; c += 3;
            if (k >= 9) { k -= 9; ++c; }
            c4[j] = c; k4[j] = k;
        }
    };

    __syncthreads();               // table ready
    stage(imtile[0]);
    __syncthreads();               // buffer 0 ready

    const uint4* wsq = (const uint4*)wpack;
    const int bkbase = (lane >> 4) * 8;
    const int bn     = lane & 15;

    for (int kc = 0; kc < NCHUNK; ++kc) {
        const int cur = kc & 1;
        // prefetch next chunk's im2col into the other buffer (overlaps WMMA)
        if (kc + 1 < NCHUNK) stage(imtile[cur ^ 1]);

        // A fragment: 2x coalesced b128 from pre-packed weights (L2-resident)
        union { v16h v; uint4 q[2]; } afrag;
        size_t abase = ((size_t)(kc * 16 + wv) * 32 + lane) * 2;
        afrag.q[0] = wsq[abase];
        afrag.q[1] = wsq[abase + 1];

        const unsigned* im32 = (const unsigned*)imtile[cur];
        #pragma unroll
        for (int j = 0; j < 4; ++j) {
            union { v16h v; unsigned u[8]; } bfrag;
            int brow = j * 16 + bn;
            #pragma unroll
            for (int d = 0; d < 8; ++d)
                bfrag.u[d] = im32[(brow * KCH + bkbase + 2 * d + ((d >> 2) << 3)) >> 1];
            acc[j] = __builtin_amdgcn_wmma_f32_16x16x32_f16(
                false, afrag.v, false, bfrag.v, (short)0, acc[j], false, false);
        }
        __syncthreads();   // next buffer written + this buffer's reads done
    }

    const int ocb  = wv * 16 + 8 * (lane >> 4);
    const int wcol = lane & 15;
    #pragma unroll
    for (int j = 0; j < 4; ++j) {
        int wp = j * 16 + wcol;
        #pragma unroll
        for (int v = 0; v < 8; ++v) {
            out[(((size_t)n * OUTC + (ocb + v)) * HDIM + h) * WDIM + wp] = acc[j][v];
        }
    }
}

// ---------------------------------------------------------------------------
// Fallback (no workspace): self-contained version with LDS weight tile.
// ---------------------------------------------------------------------------
__global__ __launch_bounds__(512)
void dcnv2_fused_wmma_fb(const float* __restrict__ x,
                         const float* __restrict__ offset,
                         const float* __restrict__ mask,
                         const float* __restrict__ weight,
                         float* __restrict__ out)
{
    __shared__ _Float16 wtile[OUTC * KCH];        // 16 KB
    __shared__ _Float16 imtile[PXW * KCH];        //  4 KB
    __shared__ uint4    tofs[PXW * KTAPS];
    __shared__ float4   twt4[PXW * KTAPS];

    const int tid  = threadIdx.x;
    const int row  = blockIdx.x;
    const int n    = row >> 6;
    const int h    = row & 63;
    const int lane = tid & 31;
    const int wv   = tid >> 5;

    build_table(tid, n, h, offset, mask, tofs, twt4);
    __syncthreads();

    v8f acc[4];
    #pragma unroll
    for (int j = 0; j < 4; ++j)
        #pragma unroll
        for (int v = 0; v < 8; ++v) acc[j][v] = 0.f;

    const char* xbase = (const char*)(x + (size_t)n * C_IN * HDIM * WDIM);
    const int ipx  = tid & 63;
    const int il0  = tid >> 6;
    const int wil  = lane;
    const int woc0 = wv;

    int c4[4], k4[4];
    #pragma unroll
    for (int j = 0; j < 4; ++j) {
        int il = il0 + j * 8;
        c4[j] = il / 9;
        k4[j] = il - c4[j] * 9;
    }

    const unsigned* wt32 = (const unsigned*)wtile;
    const unsigned* im32 = (const unsigned*)imtile;
    const int khalf  = (lane >> 4) * 8;
    const int arow   = wv * 16 + (lane & 15);
    const int bkbase = (lane >> 4) * 8;
    const int bn     = lane & 15;

    for (int kc = 0; kc < NCHUNK; ++kc) {
        const int i0 = kc * KCH;
        #pragma unroll
        for (int j = 0; j < 16; ++j) {
            int oc = woc0 + j * 16;
            wtile[oc * KCH + wil] = (_Float16)weight[(size_t)oc * KDIM + i0 + wil];
        }
        #pragma unroll
        for (int j = 0; j < 4; ++j) {
            int il = il0 + j * 8;
            int c  = c4[j];
            int k  = k4[j];
            int e  = ipx * KTAPS + k;
            uint4  o = tofs[e];
            float4 w = twt4[e];
            int cbytes = __builtin_amdgcn_readfirstlane(c << 14);
            const char* xc = xbase + (unsigned)cbytes;
            float g00 = *(const float*)(xc + o.x);
            float g01 = *(const float*)(xc + o.y);
            float g10 = *(const float*)(xc + o.z);
            float g11 = *(const float*)(xc + o.w);
            float val = w.x * g00 + w.y * g01 + w.z * g10 + w.w * g11;
            imtile[ipx * KCH + il] = (_Float16)val;
            k += 5; c += 3;
            if (k >= 9) { k -= 9; ++c; }
            c4[j] = c; k4[j] = k;
        }
        __syncthreads();

        union { v16h v; unsigned u[8]; } afrag;
        #pragma unroll
        for (int d = 0; d < 8; ++d) {
            int K = (d < 4) ? (khalf + 2 * d) : (khalf + 16 + 2 * (d - 4));
            afrag.u[d] = wt32[(arow * KCH + K) >> 1];
        }
        #pragma unroll
        for (int j = 0; j < 4; ++j) {
            union { v16h v; unsigned u[8]; } bfrag;
            int brow = j * 16 + bn;
            #pragma unroll
            for (int d = 0; d < 8; ++d)
                bfrag.u[d] = im32[(brow * KCH + bkbase + 2 * d + ((d >> 2) << 3)) >> 1];
            acc[j] = __builtin_amdgcn_wmma_f32_16x16x32_f16(
                false, afrag.v, false, bfrag.v, (short)0, acc[j], false, false);
        }
        __syncthreads();
    }

    const int ocb  = wv * 16 + 8 * (lane >> 4);
    const int wcol = lane & 15;
    #pragma unroll
    for (int j = 0; j < 4; ++j) {
        int wp = j * 16 + wcol;
        #pragma unroll
        for (int v = 0; v < 8; ++v) {
            out[(((size_t)n * OUTC + (ocb + v)) * HDIM + h) * WDIM + wp] = acc[j][v];
        }
    }
}

extern "C" void kernel_launch(void* const* d_in, const int* in_sizes, int n_in,
                              void* d_out, int out_size, void* d_ws, size_t ws_size,
                              hipStream_t stream) {
    const float* x      = (const float*)d_in[0];
    const float* offset = (const float*)d_in[1];
    const float* mask   = (const float*)d_in[2];
    const float* weight = (const float*)d_in[3];
    float* out = (float*)d_out;

    if (ws_size >= WS_BYTES && d_ws != nullptr) {
        unsigned* wpack = (unsigned*)d_ws;
        int triples = NCHUNK * 16 * 32;
        dcn_prepack_weight<<<(triples + 255) / 256, 256, 0, stream>>>(weight, wpack);
        dcnv2_fused_wmma_pk<<<dim3(8 * HDIM), dim3(512), 0, stream>>>(
            x, offset, mask, wpack, out);
    } else {
        dcnv2_fused_wmma_fb<<<dim3(8 * HDIM), dim3(512), 0, stream>>>(
            x, offset, mask, weight, out);
    }
}
